// QConv2dFast_56856777065113
// MI455X (gfx1250) — matrix operand order repaired and checked by reading, not compiled, not run on hardware
//
#include <hip/hip_runtime.h>
#include <hip/hip_bf16.h>
#include <math.h>

// ---------------------------------------------------------------------------
// QConv2d "quantum conv": out = clip(|psi @ U_sel^T|^2 * 128 / ||psi||^2, 0, 1)
// Bandwidth-bound (~38MB HBM, ~8.6 GFLOP) -> f16 WMMA GEMM, fp32 accum.
// ---------------------------------------------------------------------------

typedef __attribute__((ext_vector_type(16))) _Float16 v16h;
typedef __attribute__((ext_vector_type(8)))  float    v8f;
typedef __attribute__((ext_vector_type(4)))  int      v4i;

#define NW   8
#define QDEP 2
#define DIMQ 256

#if defined(__has_builtin)
#  if __has_builtin(__builtin_amdgcn_global_load_async_to_lds_b128)
#    define QC_ASYNC_LDS 1
#  endif
#endif

// ---------------------------------------------------------------------------
// Kernel 1: build one column of the 256x256 unitary per block; write the
// selected rows (even rows, real & imag) as pre-packed WMMA-B f16 fragments.
// B[k][n] = Re(U[2n, k])        for n in [0,128)
// B[k][n] = Im(U[2(n-128), k])  for n in [128,256)
// Fragment layout used here for v_wmma_f32_16x16x32_f16 (B is 32(K) x 16(N)), contiguous K as in A:
//   lane L: N = L%16, K group g = L/16 -> K in [16g, 16g+16); half h = K-16g.
// Packed linear index: (((kt*16 + nt)*32 + lane) * 16 + h) halves.
// ---------------------------------------------------------------------------
__global__ __launch_bounds__(128) void qc_build_u(const float* __restrict__ wts,
                                                  _Float16* __restrict__ Bpack) {
  __shared__ float2 st[DIMQ];
  __shared__ float  wq[QDEP * NW * 3];
  const int t = threadIdx.x;   // 0..127
  const int c = blockIdx.x;    // column (= K index) 0..255

  if (t < QDEP * NW * 3) wq[t] = 3.14159265358979323846f * tanhf(wts[t]);
  st[t]       = make_float2(t == c ? 1.f : 0.f, 0.f);
  st[t + 128] = make_float2((t + 128) == c ? 1.f : 0.f, 0.f);
  __syncthreads();

  for (int l = 0; l < QDEP; ++l) {
    // --- single-qubit Rot(phi,theta,omega) on every wire ---
    for (int i = 0; i < NW; ++i) {
      const float phi = wq[(l * NW + i) * 3 + 0];
      const float th  = wq[(l * NW + i) * 3 + 1];
      const float om  = wq[(l * NW + i) * 3 + 2];
      const float ch = cosf(0.5f * th), sh = sinf(0.5f * th);
      const float ph = 0.5f * (phi + om), pm = 0.5f * (phi - om);
      const float cp = cosf(ph), sp = sinf(ph);
      const float cm = cosf(pm), sm = sinf(pm);
      const int bit = 7 - i;                       // wire 0 = MSB
      const int lo  = ((t >> bit) << (bit + 1)) | (t & ((1 << bit) - 1));
      const int hi  = lo | (1 << bit);
      const float2 a0 = st[lo], a1 = st[hi];
      float2 n0, n1;
      // g00=(cp*ch,-sp*ch) g01=(-cm*sh,-sm*sh) g10=(cm*sh,-sm*sh) g11=(cp*ch,sp*ch)
      n0.x =  cp*ch*a0.x + sp*ch*a0.y - cm*sh*a1.x + sm*sh*a1.y;
      n0.y =  cp*ch*a0.y - sp*ch*a0.x - cm*sh*a1.y - sm*sh*a1.x;
      n1.x =  cm*sh*a0.x + sm*sh*a0.y + cp*ch*a1.x - sp*ch*a1.y;
      n1.y =  cm*sh*a0.y - sm*sh*a0.x + cp*ch*a1.y + sp*ch*a1.x;
      st[lo] = n0; st[hi] = n1;
      __syncthreads();
    }
    // --- entangling CNOT ring with range r ---
    const int r = (l % (NW - 1)) + 1;
    for (int i = 0; i < NW; ++i) {
      const int cbit = 7 - i;
      const int tbit = 7 - ((i + r) & 7);
      const int lo = ((t >> tbit) << (tbit + 1)) | (t & ((1 << tbit) - 1));
      const int hi = lo | (1 << tbit);
      if ((lo >> cbit) & 1) {
        const float2 tmp = st[lo]; st[lo] = st[hi]; st[hi] = tmp;
      }
      __syncthreads();
    }
  }

  // st[row] = U[row, c]; pack two values (real for n=t, imag for n=t+128)
  const float2 amp = st[2 * t];
  const int kt = c >> 5, kin = c & 31;
  const int laneBase = (kin >> 4) << 4;
  const int h = kin & 15;
  {
    const int nt = t >> 4, nl = t & 15;
    Bpack[(((kt * 16 + nt) * 32 + (laneBase + nl)) << 4) + h] = (_Float16)amp.x;
  }
  {
    const int nf = t + 128;
    const int nt = nf >> 4, nl = nf & 15;
    Bpack[(((kt * 16 + nt) * 32 + (laneBase + nl)) << 4) + h] = (_Float16)amp.y;
  }
}

// ---------------------------------------------------------------------------
// Kernel 2: fused feature-build + WMMA GEMM + epilogue.
// Grid: 512 WGs x 256 thr (8 waves). WG owns 128 consecutive patch rows
// (= 2 image rows of one batch). Wave w owns the 16-row M tile w.
// LDS: A fragments (128x256 f16, 64KB) + B fragments (256x256 f16, 128KB).
// GEMM is tiled so only one (re,im) accumulator pair is live at a time,
// keeping the wave under 256 VGPRs (no s_set_vgpr_msb churn).
// ---------------------------------------------------------------------------
__global__ __launch_bounds__(256) void qc_gemm(const float* __restrict__ xin,
                                               const _Float16* __restrict__ Bpack,
                                               float* __restrict__ out) {
  __shared__ __align__(32) _Float16 sA[128 * 256];
  __shared__ __align__(32) _Float16 sB[256 * 256];
  __shared__ float sInv[128];
  __shared__ float sPart[256];

  const int tid   = threadIdx.x;
  const int nbase = blockIdx.x * 128;

  // ---- Phase 1: B (pre-packed fragments) global -> LDS -------------------
#ifdef QC_ASYNC_LDS
  {
    const char* gsrc = (const char*)Bpack;
    char*       ldst = (char*)sB;
    for (int i = tid; i < 8192; i += 256) {   // 8192 x 16B = 128KB
      __builtin_amdgcn_global_load_async_to_lds_b128(
          (__attribute__((address_space(1))) v4i*)(gsrc + i * 16),
          (__attribute__((address_space(3))) v4i*)(ldst + i * 16),
          0, 0);
    }
#  if __has_builtin(__builtin_amdgcn_s_wait_asynccnt)
    __builtin_amdgcn_s_wait_asynccnt(0);
#  else
    asm volatile("s_wait_asynccnt 0" ::: "memory");
#  endif
  }
#else
  {
    const uint4* src = (const uint4*)Bpack;
    uint4*       dst = (uint4*)sB;
    for (int i = tid; i < 8192; i += 256) dst[i] = src[i];
  }
#endif

  // ---- Phase 2: build A fragments + row norms ----------------------------
  // thread pair (2 per row): row m = tid>>1, K half = (tid&1)*128
  {
    const int m  = tid >> 1;
    const int kh = (tid & 1) * 128;
    const int ng = nbase + m;
    const int b = ng >> 12, y = (ng >> 6) & 63, xc = ng & 63;
    float ss = 0.f;
    for (int k = kh; k < kh + 128; ++k) {
      float feat;
      if (k < 144) {
        const int cidx = k / 9, kp = k % 9, ki = kp / 3, kj = kp % 3;
        const int yy = y + ki - 1, xx = xc + kj - 1;
        float v = 0.f;
        if ((unsigned)yy < 64u && (unsigned)xx < 64u)
          v = xin[((b * 16 + cidx) * 64 + yy) * 64 + xx];
        feat = v + 0.1f;
      } else {
        feat = 0.5f;                      // AmplitudeEmbedding pad_with=0.5
      }
      ss += feat * feat;
      // A-fragment order (16b A 16x32): lane group g = kin>>4,
      // pos = kin&15; linear: ((m*8+kt)*32 + g*16 + pos)
      const int kt  = k >> 5, kin = k & 31;
      const int g   = (kin >> 4) & 1;
      const int pos = kin & 15;
      sA[(m * 8 + kt) * 32 + g * 16 + pos] = (_Float16)feat;
    }
    sPart[tid] = ss;
  }
  __syncthreads();
  if (tid < 128) sInv[tid] = 128.0f / (sPart[2 * tid] + sPart[2 * tid + 1]);
  __syncthreads();

  // ---- Phase 3: WMMA GEMM  M=16/wave, N=256, K=256 -----------------------
  const int lane = tid & 31, wave = tid >> 5;
  const int g  = lane >> 4;        // K-half group for A, C/D row group
  const int nl = lane & 15;        // N within tile / M within tile (A)
  const int mrow = wave * 16 + nl;

  // One (re,im) accumulator pair live at a time -> low VGPR pressure.
#pragma unroll
  for (int ntp = 0; ntp < 8; ++ntp) {
    v8f accRe = {};
    v8f accIm = {};
#pragma unroll
    for (int kt = 0; kt < 8; ++kt) {
      const v16h va   = *(const v16h*)&sA[(mrow * 8 + kt) * 32 + g * 16];
      const v16h vbRe = *(const v16h*)&sB[((kt * 16 + ntp) * 32 + lane) << 4];
      const v16h vbIm = *(const v16h*)&sB[((kt * 16 + ntp + 8) * 32 + lane) << 4];
      accRe = __builtin_amdgcn_wmma_f32_16x16x32_f16(
          false, va, false, vbRe, (short)0, accRe, false, false);
      accIm = __builtin_amdgcn_wmma_f32_16x16x32_f16(
          false, va, false, vbIm, (short)0, accIm, false, false);
    }
    // ---- Epilogue for this N-tile: prob, clip, NCHW scatter --------------
#pragma unroll
    for (int v = 0; v < 8; ++v) {
      const float re = accRe[v];
      const float im = accIm[v];
      const int mm = wave * 16 + v + g * 8;          // C/D layout: M = v + 8*(lane/16)
      float p = (re * re + im * im) * sInv[mm];
      p = fminf(p, 1.0f);
      const int j  = ntp * 16 + nl;                  // output channel
      const int ng = nbase + mm;
      const int b = ng >> 12, y = (ng >> 6) & 63, xc = ng & 63;
      out[((b * 128 + j) * 64 + y) * 64 + xc] = p;
    }
  }
}

extern "C" void kernel_launch(void* const* d_in, const int* in_sizes, int n_in,
                              void* d_out, int out_size, void* d_ws, size_t ws_size,
                              hipStream_t stream) {
  (void)in_sizes; (void)n_in; (void)out_size; (void)ws_size;
  const float* x   = (const float*)d_in[0];     // (16,16,64,64) f32
  const float* wts = (const float*)d_in[1];     // (2,8,3) f32
  _Float16* Bpack  = (_Float16*)d_ws;           // 256*256 f16 = 128KB scratch
  float* out       = (float*)d_out;             // (16,128,64,64) f32

  qc_build_u<<<256, 128, 0, stream>>>(wts, Bpack);
  qc_gemm<<<512, 256, 0, stream>>>(x, Bpack, out);
}
